// OctFormerBlock_29274497089728
// MI455X (gfx1250) — compile-verified, matrix operand order checked by reading, not run on hardware
//
#include <hip/hip_runtime.h>

// ---------------------------------------------------------------------------
// Types for CDNA5 WMMA (wave32): 16x16x32 bf16 -> f32 accumulate
// ---------------------------------------------------------------------------
typedef __attribute__((ext_vector_type(16))) __bf16 v16bf;
typedef __attribute__((ext_vector_type(8)))  __bf16 v8bf;
typedef __attribute__((ext_vector_type(8)))  float  v8f;

#define NPTS   131072
#define CDIM   256
#define KWIN   32
#define NWIN   4096
#define HEADS  8
#define HDIM   32
#define PBND   25
#define RPEN   51          // 2*PBND+1
#define QSCALE 0.17677669529663689f   // 32^-0.5

static __device__ __forceinline__ v16bf cat8(v8bf lo, v8bf hi) {
  return __builtin_shufflevector(lo, hi, 0,1,2,3,4,5,6,7,8,9,10,11,12,13,14,15);
}

static __device__ __forceinline__ v8f wmma_bf16(v16bf a, v16bf b, v8f c) {
  // D = A(16x32) * B(32x16) + C, fp32 accumulate
  return __builtin_amdgcn_wmma_f32_16x16x32_bf16(false, a, false, b, (short)0, c,
                                                 false, false);
}

// Async global->LDS copy of 16B, tracked by ASYNCcnt (gfx1250).
// VDST = VGPR holding wave-relative LDS byte offset, ADDR = 64-bit global addr.
static __device__ __forceinline__ void async_copy_b128(const void* gptr, void* lptr) {
  unsigned int lds_off = (unsigned int)(unsigned long long)lptr; // low 32b == LDS offset
  asm volatile("global_load_async_to_lds_b128 %0, %1, off"
               :
               : "v"(lds_off), "v"(gptr)
               : "memory");
}

static __device__ __forceinline__ void async_wait0() {
  asm volatile("s_wait_asynccnt 0x0" ::: "memory");
}

// A-fragment loader: row-major bf16 matrix, tile origin `base`, leading dim ld.
// Lane l (0..15): row=l, K elems {0..7,16..23}; lane l+16: row=l, K {8..15,24..31}.
static __device__ __forceinline__ v16bf load_afrag(const __bf16* base, int ld, int lane) {
  int row  = lane & 15;
  int koff = (lane & 16) ? 8 : 0;
  const __bf16* p = base + (size_t)row * ld + koff;
  v8bf lo = *(const v8bf*)(p);
  v8bf hi = *(const v8bf*)(p + 16);
  return cat8(lo, hi);
}

// ---------------------------------------------------------------------------
// Weight pack: fp32 W[K][Cout] (row major) -> bf16 B-fragments, fragment order
// [n16][kt][lane][16 elems] so GEMM loads one contiguous 32B vector per lane.
// ---------------------------------------------------------------------------
__global__ void pack_w_kernel(const float* __restrict__ W, __bf16* __restrict__ out,
                              int Kdim, int Cout) {
  int tid   = blockIdx.x * blockDim.x + threadIdx.x;
  int ktcnt = Kdim >> 5;
  int total = (Cout >> 4) * ktcnt * 32;
  if (tid >= total) return;
  int lane = tid & 31;
  int frag = tid >> 5;
  int kt   = frag % ktcnt;
  int n16  = frag / ktcnt;
  int n    = n16 * 16 + (lane & 15);
  int kadd = (lane & 16) ? 8 : 0;
  size_t obase = ((size_t)frag * 32 + lane) * 16;
#pragma unroll
  for (int e = 0; e < 16; ++e) {
    int k = kt * 32 + ((e < 8) ? e : e + 8) + kadd;
    out[obase + e] = (__bf16)W[(size_t)k * Cout + n];
  }
}

// ---------------------------------------------------------------------------
// CPE: x[n][c] = data[n][c] + (sum_j data[nbr[n][j]][c] * w_cpe[j][c])*g + b
// ---------------------------------------------------------------------------
__global__ void cpe_kernel(const float* __restrict__ data, const int* __restrict__ nbr,
                           const float* __restrict__ wcpe, const float* __restrict__ bg,
                           const float* __restrict__ bb, float* __restrict__ x) {
  size_t n = blockIdx.x;
  int c = threadIdx.x;
  __shared__ int nb[27];
  if (threadIdx.x < 27) nb[threadIdx.x] = nbr[n * 27 + threadIdx.x];
  __syncthreads();
  float acc = 0.f;
#pragma unroll
  for (int j = 0; j < 27; ++j)
    acc += data[(size_t)nb[j] * CDIM + c] * wcpe[j * CDIM + c];
  x[n * CDIM + c] = data[n * CDIM + c] + acc * bg[c] + bb[c];
}

// ---------------------------------------------------------------------------
// LayerNorm over C=256 (one block per row), fp32 in -> bf16 out
// ---------------------------------------------------------------------------
__global__ void ln_kernel(const float* __restrict__ x, const float* __restrict__ g,
                          const float* __restrict__ b, __bf16* __restrict__ out) {
  size_t n = blockIdx.x;
  int c = threadIdx.x;
  float v = x[n * CDIM + c];
  float s = v, s2 = v * v;
#pragma unroll
  for (int off = 16; off > 0; off >>= 1) {
    s  += __shfl_xor(s,  off, 32);
    s2 += __shfl_xor(s2, off, 32);
  }
  __shared__ float sh[8], sh2[8];
  int wv = threadIdx.x >> 5, ln = threadIdx.x & 31;
  if (ln == 0) { sh[wv] = s; sh2[wv] = s2; }
  __syncthreads();
  if (threadIdx.x < 32) {
    float a  = (ln < 8) ? sh[ln]  : 0.f;
    float a2 = (ln < 8) ? sh2[ln] : 0.f;
#pragma unroll
    for (int off = 4; off > 0; off >>= 1) {
      a  += __shfl_xor(a,  off, 32);
      a2 += __shfl_xor(a2, off, 32);
    }
    if (ln == 0) { sh[0] = a; sh2[0] = a2; }
  }
  __syncthreads();
  float mean = sh[0] * (1.f / CDIM);
  float var  = sh2[0] * (1.f / CDIM) - mean * mean;
  float r = rsqrtf(var + 1e-5f);
  out[n * CDIM + c] = (__bf16)((v - mean) * r * g[c] + b[c]);
}

// ---------------------------------------------------------------------------
// Tiled WMMA GEMM with async LDS staging of A.
// Block = 8 waves covering a 32x256 output tile (2 M-tiles x 4 strips of 64).
// A tile (32 rows x 128 K-chunk) is async-copied to LDS once per block and
// shared by all 8 waves; B comes from pre-packed fragment-order weights.
// Epilogue modes:
//  0: qkv  -> +bias, scale q cols (<256) by 32^-0.5, store bf16
//  1: proj -> +bias + residual(res), store fp32 (in-place over res allowed)
//  2: fc1  -> +bias, exact GELU, store bf16
//  3: fc2  -> +bias + residual(res), store fp32 (final output)
// ---------------------------------------------------------------------------
#define KC      128             // K chunk staged in LDS
#define LDA_SH  136             // padded LDS row stride (elems): 272B, 16B-aligned

__global__ void gemm_kernel(const __bf16* __restrict__ A, const __bf16* __restrict__ Bp,
                            const float* __restrict__ bias, const float* __restrict__ res,
                            void* __restrict__ outp, int Kdim, int Cout, int mode) {
  __shared__ __bf16 ash[32][LDA_SH];   // 8704 B

  int cblocks = Cout >> 8;                 // column blocks of 256
  int mb = blockIdx.x / cblocks;           // 32-row block
  int cb = blockIdx.x % cblocks;
  int wv   = threadIdx.x >> 5;
  int lane = threadIdx.x & 31;
  int mt_local = wv >> 2;                  // 0..1
  int n64      = cb * 4 + (wv & 3);        // 64-col strip
  int row0     = mb * 32 + mt_local * 16;
  int ktcnt    = Kdim >> 5;
  int chunks   = Kdim / KC;

  v8f zero = {0.f, 0.f, 0.f, 0.f, 0.f, 0.f, 0.f, 0.f};
  v8f acc[4] = {zero, zero, zero, zero};

  int koff  = (lane & 16) ? 8 : 0;
  int arow  = mt_local * 16 + (lane & 15);

  for (int kc = 0; kc < chunks; ++kc) {
    // ---- async stage A chunk: 32 rows x 128 cols bf16 = 512 x 16B pieces --
#pragma unroll
    for (int i = 0; i < 2; ++i) {
      int pid  = threadIdx.x + i * 256;    // 0..511
      int row  = pid >> 4;                 // 0..31
      int col8 = pid & 15;                 // 16B piece within row
      const __bf16* gp = A + (size_t)(mb * 32 + row) * Kdim + kc * KC + col8 * 8;
      async_copy_b128(gp, &ash[row][col8 * 8]);
    }
    async_wait0();
    __syncthreads();

    // ---- 4 k-tiles of 32 from LDS, 4 n-tiles from packed B ---------------
#pragma unroll
    for (int ktl = 0; ktl < 4; ++ktl) {
      const __bf16* ap = &ash[arow][ktl * 32 + koff];
      v16bf afrag = cat8(*(const v8bf*)ap, *(const v8bf*)(ap + 16));
      int kt = kc * 4 + ktl;
#pragma unroll
      for (int t = 0; t < 4; ++t) {
        int nt16 = (n64 << 2) + t;
        const v16bf* bp =
            (const v16bf*)(Bp + (((size_t)nt16 * ktcnt + kt) * 32 + lane) * 16);
        acc[t] = wmma_bf16(afrag, *bp, acc[t]);
      }
    }
    __syncthreads();   // protect LDS tile before next chunk overwrites it
  }

  // Epilogue. C/D layout: reg r, lane l -> row = r + 8*(l>>4), col = l&15.
  int rbase = (lane >> 4) << 3;
  int cl = lane & 15;
#pragma unroll
  for (int t = 0; t < 4; ++t) {
    int col = ((n64 << 2) + t) * 16 + cl;
    float bv = bias[col];
#pragma unroll
    for (int r = 0; r < 8; ++r) {
      int row = row0 + rbase + r;
      float v = acc[t][r] + bv;
      if (mode == 0) {
        if (col < CDIM) v *= QSCALE;               // q scaling fused
        ((__bf16*)outp)[(size_t)row * Cout + col] = (__bf16)v;
      } else if (mode == 1) {
        size_t idx = (size_t)row * CDIM + col;
        ((float*)outp)[idx] = v + res[idx];
      } else if (mode == 2) {
        v = 0.5f * v * (1.f + erff(v * 0.70710678118654752f));  // exact GELU
        ((__bf16*)outp)[(size_t)row * Cout + col] = (__bf16)v;
      } else {
        size_t idx = (size_t)row * CDIM + col;
        ((float*)outp)[idx] = v + res[idx];
      }
    }
  }
}

// ---------------------------------------------------------------------------
// Windowed attention: one block per window (32 tokens), one wave per head.
// qk^T and attn@v both via 4x v_wmma_f32_16x16x32_bf16; softmax + RPE fused.
// ---------------------------------------------------------------------------
__global__ void attn_kernel(const __bf16* __restrict__ qkv, const int* __restrict__ rel_pos,
                            const float* __restrict__ mask, const float* __restrict__ rpt,
                            __bf16* __restrict__ attno) {
  int w    = blockIdx.x;
  int head = threadIdx.x >> 5;
  int lane = threadIdx.x & 31;
  int n0   = w * KWIN;

  __shared__ float  attnS[HEADS][32][33];   // fp32 logits
  __shared__ __bf16 vtS[HEADS][32][40];     // v transposed: [d][key], 16B-aligned rows
  // bf16 probabilities aliased over this head's fp32 region (2560B <= 4224B)
  __bf16* aS = reinterpret_cast<__bf16*>(&attnS[head][0][0]);

  int qcol = head * HDIM;
  int kcol = CDIM + head * HDIM;
  int vcol = 2 * CDIM + head * HDIM;
  int koff = (lane & 16) ? 8 : 0;

  // ---- q*scale . k^T -> 4 WMMAs -----------------------------------------
  v8f zero = {0.f, 0.f, 0.f, 0.f, 0.f, 0.f, 0.f, 0.f};
  v16bf qa[2], kb[2];
#pragma unroll
  for (int mt = 0; mt < 2; ++mt)
    qa[mt] = load_afrag(qkv + (size_t)(n0 + mt * 16) * 768 + qcol, 768, lane);
#pragma unroll
  for (int nt = 0; nt < 2; ++nt)   // B = k^T: per-lane contiguous along d
    kb[nt] = load_afrag(qkv + (size_t)(n0 + nt * 16) * 768 + kcol, 768, lane);
  v8f acc[2][2];
#pragma unroll
  for (int mt = 0; mt < 2; ++mt)
#pragma unroll
    for (int nt = 0; nt < 2; ++nt)
      acc[mt][nt] = wmma_bf16(qa[mt], kb[nt], zero);

  // ---- stage v^T into LDS (lane = key row) ------------------------------
  {
    const __bf16* vp = qkv + (size_t)(n0 + lane) * 768 + vcol;
    v8bf vv0 = *(const v8bf*)(vp);
    v8bf vv1 = *(const v8bf*)(vp + 8);
    v8bf vv2 = *(const v8bf*)(vp + 16);
    v8bf vv3 = *(const v8bf*)(vp + 24);
#pragma unroll
    for (int e = 0; e < 8; ++e) {
      vtS[head][e][lane]      = vv0[e];
      vtS[head][8 + e][lane]  = vv1[e];
      vtS[head][16 + e][lane] = vv2[e];
      vtS[head][24 + e][lane] = vv3[e];
    }
  }

  // ---- dump logits to LDS ----------------------------------------------
  int rbase = (lane >> 4) << 3;
  int cl = lane & 15;
#pragma unroll
  for (int mt = 0; mt < 2; ++mt)
#pragma unroll
    for (int nt = 0; nt < 2; ++nt)
#pragma unroll
      for (int r = 0; r < 8; ++r)
        attnS[head][mt * 16 + rbase + r][nt * 16 + cl] = acc[mt][nt][r];
  __syncthreads();

  // ---- per-row softmax with fused RPE + mask (lane = query row) ---------
  {
    int q = lane;
    const int*   rp = rel_pos + ((size_t)(w * KWIN + q) * KWIN) * 3;
    const float* mk = mask + (size_t)w * (KWIN * KWIN) + q * KWIN;
    float a[32];
    float mx = -1e30f;
#pragma unroll
    for (int c = 0; c < 32; ++c) {
      int r0 = min(max(rp[c * 3 + 0], -PBND), PBND) + PBND;
      int r1 = min(max(rp[c * 3 + 1], -PBND), PBND) + PBND + RPEN;
      int r2 = min(max(rp[c * 3 + 2], -PBND), PBND) + PBND + 2 * RPEN;
      float rpe = rpt[r0 * HEADS + head] + rpt[r1 * HEADS + head] + rpt[r2 * HEADS + head];
      float val = attnS[head][q][c] + rpe + mk[c];
      a[c] = val;
      mx = fmaxf(mx, val);
    }
    float s = 0.f;
#pragma unroll
    for (int c = 0; c < 32; ++c) { a[c] = __expf(a[c] - mx); s += a[c]; }
    float inv = 1.f / s;
#pragma unroll
    for (int c = 0; c < 32; ++c) aS[q * 40 + c] = (__bf16)(a[c] * inv);
  }
  __syncthreads();

  // ---- out = P . v  (4 WMMAs) -------------------------------------------
  v16bf pa[2], vb[2];
#pragma unroll
  for (int mt = 0; mt < 2; ++mt) {
    const __bf16* p = aS + (size_t)(mt * 16 + (lane & 15)) * 40 + koff;
    pa[mt] = cat8(*(const v8bf*)p, *(const v8bf*)(p + 16));
  }
#pragma unroll
  for (int nt = 0; nt < 2; ++nt) {
    const __bf16* p = &vtS[head][nt * 16 + (lane & 15)][koff];
    vb[nt] = cat8(*(const v8bf*)p, *(const v8bf*)(p + 16));
  }
  v8f o[2][2];
#pragma unroll
  for (int mt = 0; mt < 2; ++mt)
#pragma unroll
    for (int nt = 0; nt < 2; ++nt)
      o[mt][nt] = wmma_bf16(pa[mt], vb[nt], zero);

#pragma unroll
  for (int mt = 0; mt < 2; ++mt)
#pragma unroll
    for (int nt = 0; nt < 2; ++nt)
#pragma unroll
      for (int r = 0; r < 8; ++r) {
        int row = n0 + mt * 16 + rbase + r;
        int col = head * HDIM + nt * 16 + cl;
        attno[(size_t)row * CDIM + col] = (__bf16)o[mt][nt][r];
      }
}

// ---------------------------------------------------------------------------
// Launch
// ---------------------------------------------------------------------------
extern "C" void kernel_launch(void* const* d_in, const int* in_sizes, int n_in,
                              void* d_out, int out_size, void* d_ws, size_t ws_size,
                              hipStream_t stream) {
  const float* data    = (const float*)d_in[0];
  const int*   nbr_idx = (const int*)d_in[1];
  const int*   rel_pos = (const int*)d_in[2];
  const float* pmask   = (const float*)d_in[3];
  const float* w_cpe   = (const float*)d_in[4];
  const float* bn_g    = (const float*)d_in[5];
  const float* bn_b    = (const float*)d_in[6];
  const float* n1_g    = (const float*)d_in[7];
  const float* n1_b    = (const float*)d_in[8];
  const float* w_qkv   = (const float*)d_in[9];
  const float* b_qkv   = (const float*)d_in[10];
  const float* w_proj  = (const float*)d_in[11];
  const float* b_proj  = (const float*)d_in[12];
  const float* rpt     = (const float*)d_in[13];
  const float* n2_g    = (const float*)d_in[14];
  const float* n2_b    = (const float*)d_in[15];
  const float* w_fc1   = (const float*)d_in[16];
  const float* b_fc1   = (const float*)d_in[17];
  const float* w_fc2   = (const float*)d_in[18];
  const float* b_fc2   = (const float*)d_in[19];
  float* out = (float*)d_out;

  // Workspace layout (all 256B-aligned by construction)
  char* p = (char*)d_ws;
  size_t o = 0;
  float*  x    = (float*)(p + o);  o += (size_t)NPTS * CDIM * 4;   // residual stream
  __bf16* h    = (__bf16*)(p + o); o += (size_t)NPTS * CDIM * 2;   // LN output (reused)
  __bf16* qkvB = (__bf16*)(p + o); o += (size_t)NPTS * 768  * 2;
  __bf16* atnB = (__bf16*)(p + o); o += (size_t)NPTS * CDIM * 2;
  __bf16* gB   = (__bf16*)(p + o); o += (size_t)NPTS * 1024 * 2;
  __bf16* wqB  = (__bf16*)(p + o); o += (size_t)256 * 768  * 2;
  __bf16* wpB  = (__bf16*)(p + o); o += (size_t)256 * 256  * 2;
  __bf16* wf1B = (__bf16*)(p + o); o += (size_t)256 * 1024 * 2;
  __bf16* wf2B = (__bf16*)(p + o); o += (size_t)1024 * 256 * 2;

  // 1) pack weights into WMMA B-fragment order (bf16)
  pack_w_kernel<<<48, 256, 0, stream>>>(w_qkv, wqB, 256, 768);
  pack_w_kernel<<<16, 256, 0, stream>>>(w_proj, wpB, 256, 256);
  pack_w_kernel<<<64, 256, 0, stream>>>(w_fc1, wf1B, 256, 1024);
  pack_w_kernel<<<64, 256, 0, stream>>>(w_fc2, wf2B, 1024, 256);

  // 2) CPE + residual -> x (fp32)
  cpe_kernel<<<NPTS, 256, 0, stream>>>(data, nbr_idx, w_cpe, bn_g, bn_b, x);

  // 3) LN1 -> h (bf16)
  ln_kernel<<<NPTS, 256, 0, stream>>>(x, n1_g, n1_b, h);

  // 4) QKV GEMM (q scaled) -> qkvB (bf16).  grid = (NPTS/32) * (768/256)
  gemm_kernel<<<12288, 256, 0, stream>>>(h, wqB, b_qkv, nullptr, qkvB, 256, 768, 0);

  // 5) windowed attention -> atnB (bf16)
  attn_kernel<<<NWIN, 256, 0, stream>>>(qkvB, rel_pos, pmask, rpt, atnB);

  // 6) proj GEMM + residual -> x (fp32, in-place).  grid = 4096 * 1
  gemm_kernel<<<4096, 256, 0, stream>>>(atnB, wpB, b_proj, x, x, 256, 256, 1);

  // 7) LN2 -> h (bf16, reuse)
  ln_kernel<<<NPTS, 256, 0, stream>>>(x, n2_g, n2_b, h);

  // 8) FC1 + exact GELU -> gB (bf16).  grid = 4096 * 4
  gemm_kernel<<<16384, 256, 0, stream>>>(h, wf1B, b_fc1, nullptr, gB, 256, 1024, 2);

  // 9) FC2 + residual -> d_out (fp32).  grid = 4096 * 1
  gemm_kernel<<<4096, 256, 0, stream>>>(gB, wf2B, b_fc2, x, out, 1024, 256, 3);
}
